// Attention_80779744903426
// MI455X (gfx1250) — matrix operand order repaired and checked by reading, not compiled, not run on hardware
//
#include <hip/hip_runtime.h>
#include <hip/hip_bf16.h>

typedef _Float16 v16h __attribute__((ext_vector_type(16)));
typedef _Float16 v8h  __attribute__((ext_vector_type(8)));
typedef _Float16 v4h  __attribute__((ext_vector_type(4)));
typedef float    v8f  __attribute__((ext_vector_type(8)));

#define DIM   1024
#define NHEAD 16
#define HD    64
#define BATCH 4
#define SEQ   2048
#define TOK   (BATCH*SEQ)   // 8192
#define QSCALE 0.03125f     // 1/sqrt(1024), folded into Q

// gfx1250 async global->LDS copy (ASYNCcnt-tracked), if toolchain exposes it
#if defined(__has_builtin)
#  if __has_builtin(__builtin_amdgcn_global_load_async_to_lds_b128)
#    define HAVE_ASYNC_LDS 1
#  endif
#endif
#ifndef HAVE_ASYNC_LDS
#  define HAVE_ASYNC_LDS 0
#endif

#if HAVE_ASYNC_LDS
// builtin signature (from hipcc diagnostic): (int4 AS1* src, int4 AS3* dst, imm, imm)
typedef int i32x4 __attribute__((vector_size(16)));
typedef __attribute__((address_space(1))) i32x4 gi32x4;
typedef __attribute__((address_space(3))) i32x4 li32x4;
#endif

union AFragU { v16h v; v8h h[2]; };

// ---- WMMA fragment loaders (wave32, 16x16x32 f16 layouts per ISA 7.12.2) ----

// A matrix 16x32 (MxK): lane row = lane&15; VGPRs 0-3 hold K=h..h+7 (pairs),
// VGPRs 4-7 hold K=16+h..16+h+7, h = 0 (lanes 0-15) or 8 (lanes 16-31).
static __device__ __forceinline__ v16h frag_a(const _Float16* A, int lda, int k0) {
    int lane = threadIdx.x & 31;
    int row  = lane & 15;
    int hh   = (lane >> 4) << 3;                 // 0 or 8
    const _Float16* p = A + (size_t)row * lda + k0 + hh;
    AFragU f;
    f.h[0] = *(const v8h*)(p);
    f.h[1] = *(const v8h*)(p + 16);
    return f.v;
}

// B matrix 32x16 (KxN) supplied as BT[n][k] (N-major): lane col = lane&15;
// lane reads K=k0+h..h+7 and k0+16+h..16+h+7, h = 0 (lanes 0-15) or 8 (lanes 16-31).
static __device__ __forceinline__ v16h frag_b(const _Float16* BT, int ldb, int k0) {
    int lane = threadIdx.x & 31;
    int col  = lane & 15;
    int hh   = (lane >> 4) << 3;                 // 0 or 8
    const _Float16* p = BT + (size_t)col * ldb + k0 + hh;
    AFragU f;
    f.h[0] = *(const v8h*)(p);
    f.h[1] = *(const v8h*)(p + 16);
    return f.v;
}

static __device__ __forceinline__ v8f wmma_f16(v16h a, v16h b, v8f c) {
    return __builtin_amdgcn_wmma_f32_16x16x32_f16(false, a, false, b, (short)0, c, false, false);
}

// ---------------- precision / layout conversion ----------------

// fp32 -> f16, 4 elements per thread (b128 in, b64 out)
__global__ void k_cvt4(const float4* __restrict__ src, v4h* __restrict__ dst, int n4) {
    int i = blockIdx.x * blockDim.x + threadIdx.x;
    if (i < n4) {
        float4 v = src[i];
        v4h h; h[0] = (_Float16)v.x; h[1] = (_Float16)v.y;
               h[2] = (_Float16)v.z; h[3] = (_Float16)v.w;
        dst[i] = h;
    }
}

// Tiled transpose+convert: src [K][Ncols] fp32 row-major -> dst [Ncols][K] f16.
// 32x32 tile via LDS; both global accesses coalesced.
__global__ __launch_bounds__(256) void k_cvt_t(const float* __restrict__ src,
                                               _Float16* __restrict__ dst,
                                               int K, int Ncols) {
    __shared__ _Float16 tile[32][33];
    int n0 = blockIdx.x * 32;                     // col block in src
    int k0 = blockIdx.y * 32;                     // row block in src
    int tx = threadIdx.x & 31, ty = threadIdx.x >> 5;   // 32 x 8
    #pragma unroll
    for (int i = 0; i < 32; i += 8)
        tile[ty + i][tx] = (_Float16)src[(size_t)(k0 + ty + i) * Ncols + n0 + tx];
    __syncthreads();
    #pragma unroll
    for (int i = 0; i < 32; i += 8)
        dst[(size_t)(n0 + ty + i) * K + k0 + tx] = tile[tx][ty + i];
}

// ---------------- QKV projection: [8192,1024] @ [1024,3072] ----------------
// Register-blocked: wave computes 64x64 (4x4 tiles of 16x16); block = 8 waves
// (2 M-waves x 4 N-waves) -> 128x256 per block. 16 WMMA per 32-K step.
// Output scattered to Q/K/V [b,h,n,d] f16; Q pre-scaled by 1/sqrt(dim).
__global__ __launch_bounds__(256) void k_qkv(const _Float16* __restrict__ xh,
                                             const _Float16* __restrict__ wT,
                                             _Float16* __restrict__ Qb,
                                             _Float16* __restrict__ Kb,
                                             _Float16* __restrict__ Vb) {
    int wid   = threadIdx.x >> 5;
    int wn    = wid & 3, wm = wid >> 2;
    int nbase = blockIdx.x * 256 + wn * 64;       // N in 0..3071
    int mbase = blockIdx.y * 128 + wm * 64;       // M in 0..8191
    const _Float16* A0 = xh + (size_t)mbase * DIM;
    const _Float16* B0 = wT + (size_t)nbase * DIM;

    v8f acc[4][4] = {};
    for (int k0 = 0; k0 < DIM; k0 += 32) {
        v16h a[4], b[4];
        #pragma unroll
        for (int mt = 0; mt < 4; mt++) a[mt] = frag_a(A0 + (size_t)mt * 16 * DIM, DIM, k0);
        #pragma unroll
        for (int nt = 0; nt < 4; nt++) b[nt] = frag_b(B0 + (size_t)nt * 16 * DIM, DIM, k0);
        #pragma unroll
        for (int mt = 0; mt < 4; mt++)
            #pragma unroll
            for (int nt = 0; nt < 4; nt++)
                acc[mt][nt] = wmma_f16(a[mt], b[nt], acc[mt][nt]);
    }

    int lane  = threadIdx.x & 31;
    int rbase = (lane >> 4) << 3;
    #pragma unroll
    for (int nt = 0; nt < 4; nt++) {
        int col   = nbase + nt * 16 + (lane & 15);
        int which = col >> 10;                    // uniform per (wave,nt)
        int rem   = col & 1023;
        int head  = rem >> 6;
        int dc    = rem & 63;
        _Float16* dst = (which == 0) ? Qb : (which == 1 ? Kb : Vb);
        float sc = (which == 0) ? QSCALE : 1.0f;
        #pragma unroll
        for (int mt = 0; mt < 4; mt++) {
            #pragma unroll
            for (int r = 0; r < 8; r++) {
                int t  = mbase + mt * 16 + rbase + r;
                int bi = t >> 11, ni = t & 2047;
                dst[(((size_t)bi * NHEAD + head) * SEQ + ni) * HD + dc] =
                    (_Float16)(acc[mt][nt][r] * sc);
            }
        }
    }
}

// ---------------- Flash attention ----------------
// grid (32, 64): blockIdx.x = 64-query block, blockIdx.y = (b*16+h).
// 128 threads = 4 waves, wave handles 16 query rows x full head dim.
// K chunk staged via async global->LDS (ASYNCcnt) when available; V chunk
// transposed through VGPRs (async path cannot transpose).
#define BQ  64
#define BK  64
#define LDK 72   // padded row stride (halves), 144B = multiple of 16B

__global__ __launch_bounds__(128) void k_attn(const _Float16* __restrict__ Qb,
                                              const _Float16* __restrict__ Kb,
                                              const _Float16* __restrict__ Vb,
                                              _Float16* __restrict__ AO) {
    __shared__ __align__(16) _Float16 Ksh[BK * LDK];      // K chunk row-major
    __shared__ __align__(16) _Float16 Vsh[HD * LDK];      // V chunk transposed [d][j]
    __shared__ __align__(16) _Float16 Psh[4 * 16 * BK];   // per-wave P staging

    const int qblock = blockIdx.x;
    const int bh     = blockIdx.y;
    const int bi     = bh >> 4, head = bh & 15;
    const int wid    = threadIdx.x >> 5;
    const int lane   = threadIdx.x & 31;

    const _Float16* Qg = Qb + (size_t)bh * SEQ * HD;
    const _Float16* Kg = Kb + (size_t)bh * SEQ * HD;
    const _Float16* Vg = Vb + (size_t)bh * SEQ * HD;

    const int qbase = qblock * BQ + wid * 16;
    v16h qf0 = frag_a(Qg + (size_t)qbase * HD, HD, 0);
    v16h qf1 = frag_a(Qg + (size_t)qbase * HD, HD, 32);

    v8f o[4] = {};
    float m[8], l[8];
    #pragma unroll
    for (int r = 0; r < 8; r++) { m[r] = -3.0e38f; l[r] = 0.0f; }

    for (int kc = 0; kc < SEQ; kc += BK) {
        __syncthreads();                                  // prev chunk done
        // stage K (row-major, async direct-to-LDS) and V (VGPR transpose)
        for (int idx = threadIdx.x; idx < (BK * HD) / 8; idx += 128) {
            int j  = (idx * 8) >> 6;
            int d0 = (idx * 8) & 63;
#if HAVE_ASYNC_LDS
            __builtin_amdgcn_global_load_async_to_lds_b128(
                (gi32x4*)(Kg + (size_t)(kc + j) * HD + d0),
                (li32x4*)&Ksh[j * LDK + d0], 0, 0);
#else
            *(v8h*)(&Ksh[j * LDK + d0]) = *(const v8h*)(Kg + (size_t)(kc + j) * HD + d0);
#endif
            v8h vv = *(const v8h*)(Vg + (size_t)(kc + j) * HD + d0);
            #pragma unroll
            for (int e = 0; e < 8; e++) Vsh[(d0 + e) * LDK + j] = vv[e];
        }
#if HAVE_ASYNC_LDS
        asm volatile("s_wait_asynccnt 0" ::: "memory");   // K chunk landed in LDS
#endif
        __syncthreads();

        // S = Q @ K^T (scale pre-folded into Q); 4 key tiles x 2 k-steps
        v8f s[4];
        #pragma unroll
        for (int jt = 0; jt < 4; jt++) {
            v8f z = {};
            z = wmma_f16(qf0, frag_b(&Ksh[jt * 16 * LDK], LDK, 0),  z);
            z = wmma_f16(qf1, frag_b(&Ksh[jt * 16 * LDK], LDK, 32), z);
            s[jt] = z;
        }

        // online softmax: row stats per VGPR slot, reduce across 16-lane half
        float sc[8];
        #pragma unroll
        for (int r = 0; r < 8; r++) {
            float v = fmaxf(fmaxf(s[0][r], s[1][r]), fmaxf(s[2][r], s[3][r]));
            #pragma unroll
            for (int off = 1; off < 16; off <<= 1) v = fmaxf(v, __shfl_xor(v, off, 32));
            float mn = fmaxf(m[r], v);
            sc[r] = __expf(m[r] - mn);
            m[r]  = mn;
        }
        _Float16* Pw = &Psh[wid * 16 * BK];
        float rs[8];
        #pragma unroll
        for (int r = 0; r < 8; r++) rs[r] = 0.0f;
        #pragma unroll
        for (int jt = 0; jt < 4; jt++) {
            #pragma unroll
            for (int r = 0; r < 8; r++) {
                float p = __expf(s[jt][r] - m[r]);
                rs[r] += p;
                int row = ((lane >> 4) << 3) + r;
                Pw[row * BK + jt * 16 + (lane & 15)] = (_Float16)p;
            }
        }
        #pragma unroll
        for (int r = 0; r < 8; r++) {
            float v = rs[r];
            #pragma unroll
            for (int off = 1; off < 16; off <<= 1) v += __shfl_xor(v, off, 32);
            l[r] = l[r] * sc[r] + v;
        }
        #pragma unroll
        for (int dt = 0; dt < 4; dt++)
            #pragma unroll
            for (int r = 0; r < 8; r++) o[dt][r] *= sc[r];

        __syncthreads();                                  // P visible (C->A relayout)

        // O += P @ V
        v16h p0 = frag_a(Pw, BK, 0);
        v16h p1 = frag_a(Pw, BK, 32);
        #pragma unroll
        for (int dt = 0; dt < 4; dt++) {
            o[dt] = wmma_f16(p0, frag_b(&Vsh[dt * 16 * LDK], LDK, 0),  o[dt]);
            o[dt] = wmma_f16(p1, frag_b(&Vsh[dt * 16 * LDK], LDK, 32), o[dt]);
        }
    }

    // normalize and write to attn_out [token][dim] f16
    #pragma unroll
    for (int r = 0; r < 8; r++) {
        float invl = 1.0f / l[r];
        int row = ((lane >> 4) << 3) + r;
        size_t tok = (size_t)bi * SEQ + (qbase + row);
        #pragma unroll
        for (int dt = 0; dt < 4; dt++) {
            int colg = head * HD + dt * 16 + (lane & 15);
            AO[tok * DIM + colg] = (_Float16)(o[dt][r] * invl);
        }
    }
}

// ---------------- output projection: [8192,1024] @ [1024,1024] + bias ----------------
// Same 4x4 register blocking; block = 2 M-waves x 4 N-waves -> 128x256.
__global__ __launch_bounds__(256) void k_proj(const _Float16* __restrict__ AO,
                                              const _Float16* __restrict__ wT,
                                              const float* __restrict__ bias,
                                              float* __restrict__ out) {
    int wid   = threadIdx.x >> 5;
    int wn    = wid & 3, wm = wid >> 2;
    int nbase = blockIdx.x * 256 + wn * 64;       // 0..1023
    int mbase = blockIdx.y * 128 + wm * 64;       // 0..8191
    const _Float16* A0 = AO + (size_t)mbase * DIM;
    const _Float16* B0 = wT + (size_t)nbase * DIM;

    v8f acc[4][4] = {};
    for (int k0 = 0; k0 < DIM; k0 += 32) {
        v16h a[4], b[4];
        #pragma unroll
        for (int mt = 0; mt < 4; mt++) a[mt] = frag_a(A0 + (size_t)mt * 16 * DIM, DIM, k0);
        #pragma unroll
        for (int nt = 0; nt < 4; nt++) b[nt] = frag_b(B0 + (size_t)nt * 16 * DIM, DIM, k0);
        #pragma unroll
        for (int mt = 0; mt < 4; mt++)
            #pragma unroll
            for (int nt = 0; nt < 4; nt++)
                acc[mt][nt] = wmma_f16(a[mt], b[nt], acc[mt][nt]);
    }

    int lane  = threadIdx.x & 31;
    int rbase = (lane >> 4) << 3;
    #pragma unroll
    for (int nt = 0; nt < 4; nt++) {
        int col  = nbase + nt * 16 + (lane & 15);
        float bb = bias[col];
        #pragma unroll
        for (int mt = 0; mt < 4; mt++) {
            #pragma unroll
            for (int r = 0; r < 8; r++) {
                int t = mbase + mt * 16 + rbase + r;
                out[(size_t)t * DIM + col] = acc[mt][nt][r] + bb;
            }
        }
    }
}

extern "C" void kernel_launch(void* const* d_in, const int* in_sizes, int n_in,
                              void* d_out, int out_size, void* d_ws, size_t ws_size,
                              hipStream_t stream) {
    const float* x      = (const float*)d_in[0];
    const float* w_qkv  = (const float*)d_in[1];
    const float* w_proj = (const float*)d_in[2];
    const float* b_proj = (const float*)d_in[3];
    float* out = (float*)d_out;

    // workspace layout (f16), total ~92.3 MB
    _Float16* xh     = (_Float16*)d_ws;
    _Float16* wqkvT  = xh     + (size_t)TOK * DIM;
    _Float16* wprojT = wqkvT  + (size_t)3 * DIM * DIM;
    _Float16* Qb     = wprojT + (size_t)DIM * DIM;
    _Float16* Kb     = Qb     + (size_t)TOK * DIM;
    _Float16* Vb     = Kb     + (size_t)TOK * DIM;
    _Float16* AO     = Vb     + (size_t)TOK * DIM;

    int n4 = TOK * DIM / 4;
    k_cvt4<<<(n4 + 255) / 256, 256, 0, stream>>>((const float4*)x, (v4h*)xh, n4);
    k_cvt_t<<<dim3(3 * DIM / 32, DIM / 32), 256, 0, stream>>>(w_qkv, wqkvT, DIM, 3 * DIM);
    k_cvt_t<<<dim3(DIM / 32, DIM / 32), 256, 0, stream>>>(w_proj, wprojT, DIM, DIM);

    k_qkv <<<dim3(3 * DIM / 256, TOK / 128), 256, 0, stream>>>(xh, wqkvT, Qb, Kb, Vb);
    k_attn<<<dim3(SEQ / BQ, BATCH * NHEAD), 128, 0, stream>>>(Qb, Kb, Vb, AO);
    k_proj<<<dim3(DIM / 256, TOK / 128), 256, 0, stream>>>(AO, wprojT, b_proj, out);
}
